// DescriptorNet_68401649156536
// MI455X (gfx1250) — matrix operand order repaired and checked by reading, not compiled, not run on hardware
//
#include <hip/hip_runtime.h>

typedef __attribute__((ext_vector_type(16))) _Float16 v16h;
typedef __attribute__((ext_vector_type(8)))  float    v8f;

#define NEDGE   256
#define NC      64      // edges per LDS chunk
#define NCHUNK  4
#define H0      25
#define H1      50
#define H2      100
#define MOUT    100
#define MP      8

// padded dims for WMMA
#define K0P 32          // 25 -> 32  (1 k-iter)
#define N1P 64          // 50 -> 64  (4 n-tiles)
#define K1P 64          // 50 -> 64  (2 k-iters)
#define N2P 112         // 100 -> 112 (7 n-tiles)

// LDS strides (elements) — row offsets 16B-aligned, strides spread banks
#define SW1   72
#define SW2   120
#define SENVT 72
#define SACT0 40
#define SACT1 72
#define SACT2 120
#define STT   116

struct __align__(16) Smem {
  _Float16 w1[32 * SW1];        //  4608 B : W1 f16 (K=32 x N=64 used)
  _Float16 w2[64 * SW2];        // 15360 B : W2 f16 (K=64 x N=112 used)
  _Float16 envT[16 * SENVT];    //  2304 B : Env^T rows c=0..3 valid, 4..15 zero
  _Float16 act0[NC * SACT0];    //  5120 B
  _Float16 act1[NC * SACT1];    //  9216 B
  _Float16 act2[NC * SACT2];    // 15360 B
  float tt[16 * STT];           //  7424 B : T^T (rows 0..3 valid)
  float si[MOUT * 4];           //  1600 B
  float b1[H1], g1[H1], bt1[H1];
  float b2[H2], g2[H2], bt2[H2];
  float bo[MOUT];
  float w0[H0], b0[H0], g0[H0], bt0[H0];
  float pairw[36];
  float envsum[4];
};                              // ~63.7 KB < 64 KB static

__device__ __forceinline__ v8f wmma_f16(v16h a, v16h b, v8f c) {
  return __builtin_amdgcn_wmma_f32_16x16x32_f16(false, a, false, b, (short)0, c, false, false);
}

// A-fragment (16x32 f16, MxK): lane<16 -> K 0-7 & 16-23 ; lane>=16 -> K 8-15 & 24-31
__device__ __forceinline__ v16h frag_A(const _Float16* base, int stride, int row, int kIter, int lane) {
  const int k0 = kIter * 32 + ((lane & 16) ? 8 : 0);
  const _Float16* p = base + row * stride + k0;
  v16h r;
  *reinterpret_cast<uint4*>(&r)       = *reinterpret_cast<const uint4*>(p);
  *(reinterpret_cast<uint4*>(&r) + 1) = *reinterpret_cast<const uint4*>(p + 16);
  return r;
}

// B-fragment (32x16 f16, KxN): N = lane&15 ; lane<16 -> K 0-7 & 16-23 ; lane>=16 -> K 8-15 & 24-31
__device__ __forceinline__ v16h frag_B(const _Float16* base, int stride, int krow, int n0) {
  const int kb  = (krow & ~31) + ((krow & 16) ? 8 : 0);
  const int col = n0 + (krow & 15);
  v16h r;
#pragma unroll
  for (int s = 0; s < 8; ++s) {
    r[s]     = base[(kb + s) * stride + col];
    r[8 + s] = base[(kb + 16 + s) * stride + col];
  }
  return r;
}

// D (16x16 f32): VGPR r, lanes 0-15 -> M=r, lanes 16-31 -> M=r+8 ; N = lane&15
__device__ __forceinline__ void store_D16(_Float16* base, int stride, int mBase, int n0, int lane, v8f d) {
  const int col = n0 + (lane & 15);
  const int rb  = mBase + ((lane & 16) ? 8 : 0);
#pragma unroll
  for (int r = 0; r < 8; ++r) base[(rb + r) * stride + col] = (_Float16)d[r];
}

// LayerNorm+ReLU on one LDS f16 row, 4 threads per row (q = 0..3), bias folded in.
__device__ __forceinline__ void ln_relu_quad(_Float16* rowp, int q, int nvalid, int npad,
                                             const float* b, const float* g, const float* bt) {
  float sum = 0.f, sumsq = 0.f;
  for (int j = q; j < nvalid; j += 4) {
    float h = (float)rowp[j] + b[j];
    sum += h; sumsq += h * h;
  }
  sum   += __shfl_xor(sum,   1, 4);
  sum   += __shfl_xor(sum,   2, 4);
  sumsq += __shfl_xor(sumsq, 1, 4);
  sumsq += __shfl_xor(sumsq, 2, 4);
  const float mu   = sum / nvalid;
  const float var  = fmaxf(sumsq / nvalid - mu * mu, 0.f);
  const float rstd = rsqrtf(var + 1e-5f);
  for (int j = q; j < npad; j += 4) {
    _Float16 o = (_Float16)0.f;
    if (j < nvalid) {
      float h = (float)rowp[j] + b[j];
      float v = (h - mu) * rstd * g[j] + bt[j];
      o = (_Float16)fmaxf(v, 0.f);
    }
    rowp[j] = o;
  }
}

__global__ __launch_bounds__(256, 1) void descnet_kernel(
    const float* __restrict__ edge_vecs, const int* __restrict__ Zi, const int* __restrict__ Zj,
    const float* __restrict__ pair_w,
    const float* __restrict__ W0, const float* __restrict__ b0, const float* __restrict__ g0, const float* __restrict__ bt0,
    const float* __restrict__ W1, const float* __restrict__ b1, const float* __restrict__ g1, const float* __restrict__ bt1,
    const float* __restrict__ W2, const float* __restrict__ b2, const float* __restrict__ g2, const float* __restrict__ bt2,
    const float* __restrict__ Wout, const float* __restrict__ bout,
    float* __restrict__ out)
{
  __shared__ Smem sm;
  const int tid  = threadIdx.x;
  const int a    = blockIdx.x;
  const int lane = tid & 31;
  // wave id, forced wave-uniform so tile loops/guards compile to scalar branches
  // (keeps EXEC all-ones around v_wmma, no per-lane loop bookkeeping)
  const int wvu  = __builtin_amdgcn_readfirstlane(tid >> 5);

  // ---- stage weights / params into LDS (f16 conversion for WMMA operands) ----
  for (int i = tid; i < 32 * 64; i += 256) { int k = i >> 6, j = i & 63;
    sm.w1[k * SW1 + j] = (_Float16)((k < H0 && j < H1) ? W1[k * H1 + j] : 0.f); }
  for (int i = tid; i < 64 * 112; i += 256) { int k = i / 112, j = i % 112;
    sm.w2[k * SW2 + j] = (_Float16)((k < H1 && j < H2) ? W2[k * H2 + j] : 0.f); }
  for (int i = tid; i < 12 * SENVT; i += 256) sm.envT[4 * SENVT + i] = (_Float16)0.f;
  for (int i = tid; i < H1; i += 256) { sm.b1[i] = b1[i]; sm.g1[i] = g1[i]; sm.bt1[i] = bt1[i]; }
  for (int i = tid; i < H2; i += 256) { sm.b2[i] = b2[i]; sm.g2[i] = g2[i]; sm.bt2[i] = bt2[i]; sm.bo[i] = bout[i]; }
  for (int i = tid; i < H0; i += 256) { sm.w0[i] = W0[i]; sm.b0[i] = b0[i]; sm.g0[i] = g0[i]; sm.bt0[i] = bt0[i]; }
  if (tid < 36) sm.pairw[tid] = pair_w[tid];
  if (tid < 4)  sm.envsum[tid] = 0.f;
  __syncthreads();

  // ---- per-edge env + layer0 (+LN+ReLU) in registers ----
  const int n = tid;
  const int eoff = (a * NEDGE + n) * 3;
  const float ex = edge_vecs[eoff], ey = edge_vecs[eoff + 1], ez = edge_vecs[eoff + 2];
  const int zi = Zi[a * NEDGE + n], zj = Zj[a * NEDGE + n];
  const float pw = sm.pairw[zi * 6 + zj];
  const float r  = sqrtf(ex * ex + ey * ey + ez * ez);
  const float rs = fmaxf(r, 1e-8f);
  const float inv = 1.0f / rs;
  const float u = (rs - 2.0f) * 0.25f;
  const float poly = u * u * u * (-6.0f * u * u + 15.0f * u - 10.0f);
  const float sij = (rs < 2.0f) ? inv : ((rs < 6.0f) ? inv * (poly + 1.0f) : 0.0f);
  float env[4];
  env[0] = (sij - 0.05f) * 50.0f * pw;          // /S_STD
  const float kk = sij * 20.0f * pw;            // /COORDS_STD
  env[1] = ex * kk; env[2] = ey * kk; env[3] = ez * kk;
#pragma unroll
  for (int c = 0; c < 4; ++c) atomicAdd(&sm.envsum[c], env[c]);

  float h0[H0], a0[H0];
  float mu0 = 0.f;
#pragma unroll
  for (int k = 0; k < H0; ++k) { h0[k] = env[0] * sm.w0[k] + sm.b0[k]; mu0 += h0[k]; }
  mu0 *= (1.0f / H0);
  float var0 = 0.f;
#pragma unroll
  for (int k = 0; k < H0; ++k) { float d = h0[k] - mu0; var0 += d * d; }
  var0 *= (1.0f / H0);
  const float rstd0 = rsqrtf(var0 + 1e-5f);
#pragma unroll
  for (int k = 0; k < H0; ++k)
    a0[k] = fmaxf((h0[k] - mu0) * rstd0 * sm.g0[k] + sm.bt0[k], 0.f);

  // ---- chunked pipeline: L1 WMMA -> LN -> L2 WMMA -> LN -> T^T accumulation ----
  v8f accT = {0.f, 0.f, 0.f, 0.f, 0.f, 0.f, 0.f, 0.f};   // per-wave T^T tile (waves 0..6)

  for (int ch = 0; ch < NCHUNK; ++ch) {
    if ((tid >> 6) == ch) {
      const int nl = tid & 63;
#pragma unroll
      for (int c = 0; c < 4; ++c) sm.envT[c * SENVT + nl] = (_Float16)env[c];
#pragma unroll
      for (int k = 0; k < K0P; ++k) sm.act0[nl * SACT0 + k] = (_Float16)(k < H0 ? a0[k] : 0.f);
    }
    __syncthreads();

    // layer1: (64 x 32) @ (32 x 64) — 4 Mtiles x 4 Ntiles, K=32 ; 2 tiles per wave
#pragma unroll
    for (int it = 0; it < 2; ++it) {
      const int t  = wvu + it * 8;
      const int mT = t & 3, nt = t >> 2;
      v16h Af = frag_A(sm.act0, SACT0, mT * 16 + (lane & 15), 0, lane);
      v16h Bf = frag_B(sm.w1, SW1, lane, nt * 16);
      v8f d = {0.f, 0.f, 0.f, 0.f, 0.f, 0.f, 0.f, 0.f};
      d = wmma_f16(Af, Bf, d);
      store_D16(sm.act1, SACT1, mT * 16, nt * 16, lane, d);
    }
    __syncthreads();
    ln_relu_quad(&sm.act1[(tid >> 2) * SACT1], tid & 3, H1, N1P, sm.b1, sm.g1, sm.bt1);
    __syncthreads();

    // layer2: (64 x 64) @ (64 x 112) — 4 Mtiles x 7 Ntiles, K=64 ; scalar-uniform loop
    for (int t = wvu; t < 28; t += 8) {
      const int mT = t & 3, nt = t >> 2;
      v8f d = {0.f, 0.f, 0.f, 0.f, 0.f, 0.f, 0.f, 0.f};
#pragma unroll
      for (int kI = 0; kI < 2; ++kI) {
        v16h Af = frag_A(sm.act1, SACT1, mT * 16 + (lane & 15), kI, lane);
        v16h Bf = frag_B(sm.w2, SW2, lane + kI * 32, nt * 16);
        d = wmma_f16(Af, Bf, d);
      }
      store_D16(sm.act2, SACT2, mT * 16, nt * 16, lane, d);
    }
    __syncthreads();
    ln_relu_quad(&sm.act2[(tid >> 2) * SACT2], tid & 3, H2, N2P, sm.b2, sm.g2, sm.bt2);
    __syncthreads();

    // T^T += Env^T (16x64, rows 0..3 valid) @ X2 (64x112); wave w owns N-tile w
    if (wvu < 7) {
#pragma unroll
      for (int kI = 0; kI < 2; ++kI) {
        v16h Af = frag_A(sm.envT, SENVT, (lane & 15), kI, lane);
        v16h Bf = frag_B(sm.act2, SACT2, lane + kI * 32, wvu * 16);
        accT = wmma_f16(Af, Bf, accT);
      }
    }
    __syncthreads();
  }

  // ---- spill T^T to LDS ----
  if (wvu < 7) {
    const int col = wvu * 16 + (lane & 15);
    const int rb  = (lane & 16) ? 8 : 0;
#pragma unroll
    for (int r2 = 0; r2 < 8; ++r2) sm.tt[(r2 + rb) * STT + col] = accT[r2];
  }
  __syncthreads();

  // ---- Si[m,c] = sum_j Wout[j,m]*T[j,c] + bout[m]*envsum[c]  (f32 VALU) ----
  if (tid < MOUT) {
    float s0 = sm.bo[tid] * sm.envsum[0];
    float s1 = sm.bo[tid] * sm.envsum[1];
    float s2 = sm.bo[tid] * sm.envsum[2];
    float s3 = sm.bo[tid] * sm.envsum[3];
    for (int j = 0; j < H2; ++j) {
      const float w = Wout[j * MOUT + tid];          // coalesced, L2-resident
      s0 += w * sm.tt[0 * STT + j];
      s1 += w * sm.tt[1 * STT + j];
      s2 += w * sm.tt[2 * STT + j];
      s3 += w * sm.tt[3 * STT + j];
    }
    sm.si[tid * 4 + 0] = s0; sm.si[tid * 4 + 1] = s1;
    sm.si[tid * 4 + 2] = s2; sm.si[tid * 4 + 3] = s3;
  }
  __syncthreads();

  // ---- Di[m,p] = dot4(Si[m], Si[p]), p < 8 ----
  if (tid < MOUT) {
    const float m0 = sm.si[tid * 4 + 0], m1 = sm.si[tid * 4 + 1];
    const float m2 = sm.si[tid * 4 + 2], m3 = sm.si[tid * 4 + 3];
#pragma unroll
    for (int p = 0; p < MP; ++p) {
      const float d = m0 * sm.si[p * 4 + 0] + m1 * sm.si[p * 4 + 1]
                    + m2 * sm.si[p * 4 + 2] + m3 * sm.si[p * 4 + 3];
      out[a * (MOUT * MP) + tid * MP + p] = d;
    }
  }
}

extern "C" void kernel_launch(void* const* d_in, const int* in_sizes, int n_in,
                              void* d_out, int out_size, void* d_ws, size_t ws_size,
                              hipStream_t stream) {
  (void)in_sizes; (void)n_in; (void)out_size; (void)d_ws; (void)ws_size;
  const float* edge_vecs = (const float*)d_in[0];
  const int*   Zi        = (const int*)d_in[1];
  const int*   Zj        = (const int*)d_in[2];
  const float* pair_w    = (const float*)d_in[3];
  const float* W0  = (const float*)d_in[4];
  const float* b0  = (const float*)d_in[5];
  const float* g0  = (const float*)d_in[6];
  const float* bt0 = (const float*)d_in[7];
  const float* W1  = (const float*)d_in[8];
  const float* b1  = (const float*)d_in[9];
  const float* g1  = (const float*)d_in[10];
  const float* bt1 = (const float*)d_in[11];
  const float* W2  = (const float*)d_in[12];
  const float* b2  = (const float*)d_in[13];
  const float* g2  = (const float*)d_in[14];
  const float* bt2 = (const float*)d_in[15];
  const float* Wout = (const float*)d_in[16];
  const float* bout = (const float*)d_in[17];
  float* out = (float*)d_out;

  descnet_kernel<<<4096, 256, 0, stream>>>(
      edge_vecs, Zi, Zj, pair_w,
      W0, b0, g0, bt0, W1, b1, g1, bt1, W2, b2, g2, bt2,
      Wout, bout, out);
}
